// jump_ode_29025388986952
// MI455X (gfx1250) — matrix operand order repaired and checked by reading, not compiled, not run on hardware
//
#include <hip/hip_runtime.h>

// ---------------------------------------------------------------------------
// Problem constants
// ---------------------------------------------------------------------------
#define PASTLEN 4
#define BATCH   32768
#define DIM     64
#define JUMPD   16
#define HID     100
#define INDIM   256          // PASTLEN*DIM
#define HP      128          // HID padded to multiple of 32 (K and N pad)
#define JCOLS   2048         // augmented head: 1024 logits + 64*16 (zero|15 jumps)
#define ASTRIDE 136          // LDS activation row stride in halves (conflict-free, 16B aligned)
#define FSTRIDE 68           // LDS f32 row stride (drift / sel)
#define SLOT    4352         // bytes per LDS slot: 16*ASTRIDE*2 == 16*FSTRIDE*4
#define NWAVES  4
#define JOFF    (NWAVES * 4 * SLOT)   // 69632, 1KB aligned
#define CHBYTES 16384        // TDM chunk: 16 B-tiles (4 t-values)
#define NCHUNK  32           // 512KB jump head / 16KB

// ws byte offsets (all >=256B aligned)
#define OFF_W1D 0u
#define OFF_W1J 65536u
#define OFF_W2D 131072u
#define OFF_W3D 163840u
#define OFF_W2J 196608u
#define OFF_W3J 229376u
#define OFF_W4D 262144u
#define OFF_W4J 278528u
#define OFF_B1D 802816u
#define OFF_B2D 803328u
#define OFF_B3D 803840u
#define OFF_B1J 804352u
#define OFF_B2J 804864u
#define OFF_B3J 805376u
#define OFF_B4D 805888u
#define OFF_B4J 806144u
#define OFF_PART 814336u     // 512 blocks * 4 waves * 2 floats = 16384B
#define WS_NEED 830720u

typedef _Float16 v16h __attribute__((ext_vector_type(16)));
typedef _Float16 v8h  __attribute__((ext_vector_type(8)));
typedef float    v8f  __attribute__((ext_vector_type(8)));
typedef unsigned int u32x4 __attribute__((ext_vector_type(4)));
typedef int      i32x8 __attribute__((ext_vector_type(8)));
typedef int      i32x4 __attribute__((ext_vector_type(4)));

static __device__ __forceinline__ v8f wmma_f16(v16h a, v16h b, v8f c) {
  return __builtin_amdgcn_wmma_f32_16x16x32_f16(false, a, false, b, (short)0, c,
                                                false, false);
}

// B fragment (global L2 or LDS): tile = contiguous 512 halves (1KB); lane reads 32B.
static __device__ __forceinline__ v16h bfrag(const _Float16* W, int tileIdx, int lane) {
  return *(const v16h*)(W + (((size_t)tileIdx) << 9) + (lane << 4));
}

// A fragment from LDS activation buffer [16][ASTRIDE] f16.
static __device__ __forceinline__ v16h afrag_lds(const _Float16* buf, int lane, int kt) {
  const int m = lane & 15, kh = lane >> 4;
  const _Float16* r = buf + m * ASTRIDE + kt * 32 + kh * 8;
  v8h lo = *(const v8h*)r;
  v8h hi = *(const v8h*)(r + 16);
  v16h a;
#pragma unroll
  for (int i = 0; i < 8; ++i) { a[i] = lo[i]; a[i + 8] = hi[i]; }
  return a;
}

// Layer-1 A fragment straight from curstate (4,32768,64) f32; K = p*64 + d.
static __device__ __forceinline__ v16h afrag_l1(const float* __restrict__ cur,
                                                int rowbase, int lane, int kt) {
  const int m = lane & 15, kh = lane >> 4;
  const int row = rowbase + m;
  v16h a;
  {
    int kc = kt * 32 + kh * 8;
    const float4* q = (const float4*)(cur + ((size_t)(kc >> 6) * BATCH + row) * DIM + (kc & 63));
    float4 x0 = q[0], x1 = q[1];
    a[0] = (_Float16)x0.x; a[1] = (_Float16)x0.y; a[2] = (_Float16)x0.z; a[3] = (_Float16)x0.w;
    a[4] = (_Float16)x1.x; a[5] = (_Float16)x1.y; a[6] = (_Float16)x1.z; a[7] = (_Float16)x1.w;
  }
  {
    int kc = kt * 32 + 16 + kh * 8;
    const float4* q = (const float4*)(cur + ((size_t)(kc >> 6) * BATCH + row) * DIM + (kc & 63));
    float4 x0 = q[0], x1 = q[1];
    a[8]  = (_Float16)x0.x; a[9]  = (_Float16)x0.y; a[10] = (_Float16)x0.z; a[11] = (_Float16)x0.w;
    a[12] = (_Float16)x1.x; a[13] = (_Float16)x1.y; a[14] = (_Float16)x1.z; a[15] = (_Float16)x1.w;
  }
  return a;
}

static __device__ __forceinline__ void gemm_h(const _Float16* Abuf,
                                              const _Float16* __restrict__ W,
                                              int lane, v8f acc[8]) {
#pragma unroll
  for (int kt = 0; kt < 4; ++kt) {
    v16h a = afrag_lds(Abuf, lane, kt);
#pragma unroll
    for (int nt = 0; nt < 8; ++nt)
      acc[nt] = wmma_f16(a, bfrag(W, nt * 4 + kt, lane), acc[nt]);
  }
}

static __device__ __forceinline__ void store_tanh(const v8f acc[8],
                                                  const float* __restrict__ bias,
                                                  _Float16* dst, int lane) {
  const int n = lane & 15, kh = lane >> 4;
#pragma unroll
  for (int nt = 0; nt < 8; ++nt) {
    float bv = bias[nt * 16 + n];
#pragma unroll
    for (int r = 0; r < 8; ++r)
      dst[(r + 8 * kh) * ASTRIDE + nt * 16 + n] = (_Float16)tanhf(acc[nt][r] + bv);
  }
}

static __device__ __forceinline__ void store_relures(const v8f acc[8],
                                                     const float* __restrict__ bias,
                                                     const _Float16* h1,
                                                     _Float16* dst, int lane) {
  const int n = lane & 15, kh = lane >> 4;
#pragma unroll
  for (int nt = 0; nt < 8; ++nt) {
    float bv = bias[nt * 16 + n];
#pragma unroll
    for (int r = 0; r < 8; ++r) {
      int m = r + 8 * kh, c = nt * 16 + n;
      float v = acc[nt][r] + bv + (float)h1[m * ASTRIDE + c];
      dst[m * ASTRIDE + c] = (_Float16)fmaxf(v, 0.f);
    }
  }
}

// TDM: async 1-D copy of `bytes` (multiple of 8) global -> LDS. TENSORcnt tracked.
static __device__ __forceinline__ void tdm_load_1d(unsigned int lds_off,
                                                   const void* gsrc,
                                                   unsigned int bytes) {
  unsigned int n = bytes >> 3;                       // elements of 8B
  unsigned long long ga = (unsigned long long)(size_t)gsrc;
  u32x4 g0;
  g0[0] = 1u;                                        // count=1 valid descriptor
  g0[1] = lds_off;                                   // lds_addr [63:32]
  g0[2] = (unsigned int)ga;                          // global_addr [95:64]
  g0[3] = (unsigned int)(ga >> 32) | (2u << 30);     // global_addr hi + type=2
  i32x8 g1;
  g1[0] = (int)(3u << 16);                           // data_size=8B, no multicast
  g1[1] = (int)((n & 0xFFFFu) << 16);                // tensor_dim0 lo16
  g1[2] = (int)(((n >> 16) & 0xFFFFu) | (1u << 16)); // tensor_dim0 hi16 | tensor_dim1=1
  g1[3] = (int)((n & 0xFFFFu) << 16);                // tile_dim0 (n <= 65535)
  g1[4] = 1;                                         // tile_dim1=1, tile_dim2=0
  g1[5] = (int)n;                                    // tensor_dim0_stride lo32
  g1[6] = 0;
  g1[7] = 0;
  i32x4 gz4 = {0, 0, 0, 0};
  i32x8 gz8 = {0, 0, 0, 0, 0, 0, 0, 0};
  __builtin_amdgcn_tensor_load_to_lds(g0, g1, gz4, gz4, gz8, 0);
}

// ---------------------------------------------------------------------------
// Prep: fp32 row-major weights -> f16 fragment-major tiles (zero padded)
// ---------------------------------------------------------------------------
__global__ void prep_wtiles(const float* __restrict__ src, _Float16* __restrict__ dst,
                            int Kin, int Nout, int KT, int NT) {
  int idx = blockIdx.x * blockDim.x + threadIdx.x;
  int total = NT * KT * 512;
  if (idx >= total) return;
  int e = idx & 15, l = (idx >> 4) & 31, tile = idx >> 9;
  int kt = tile % KT, nt = tile / KT;
  int K = kt * 32 + (e >> 3) * 16 + (l >> 4) * 8 + (e & 7);
  int c = nt * 16 + (l & 15);
  float v = (K < Kin && c < Nout) ? src[(size_t)K * Nout + c] : 0.f;
  dst[idx] = (_Float16)v;
}

__global__ void prep_wj4(const float* __restrict__ src, _Float16* __restrict__ dst) {
  int idx = blockIdx.x * blockDim.x + threadIdx.x;           // 128*4*512 total
  if (idx >= 128 * 4 * 512) return;
  int e = idx & 15, l = (idx >> 4) & 31, tile = idx >> 9;
  int kt = tile % 4, nt = tile / 4;
  int K = kt * 32 + (e >> 3) * 16 + (l >> 4) * 8 + (e & 7);
  int gc = nt * 16 + (l & 15);
  int sc;
  if (gc < 1024) sc = gc;
  else { int d = (gc - 1024) >> 4, q = (gc - 1024) & 15; sc = q ? (1024 + d * 15 + q - 1) : -1; }
  float v = (K < HID && sc >= 0) ? src[(size_t)K * 1984 + sc] : 0.f;
  dst[idx] = (_Float16)v;
}

__global__ void prep_bias(const float* __restrict__ src, float* __restrict__ dst,
                          int Nout, int Npad) {
  int i = blockIdx.x * blockDim.x + threadIdx.x;
  if (i < Npad) dst[i] = (i < Nout) ? src[i] : 0.f;
}

__global__ void prep_bj4(const float* __restrict__ src, float* __restrict__ dst) {
  int gc = blockIdx.x * blockDim.x + threadIdx.x;
  if (gc >= JCOLS) return;
  float v;
  if (gc < 1024) v = src[gc];
  else { int d = (gc - 1024) >> 4, q = (gc - 1024) & 15; v = q ? src[1024 + d * 15 + q - 1] : 0.f; }
  dst[gc] = v;
}

// ---------------------------------------------------------------------------
// Fused main kernel: 512 blocks x 128 threads; wave owns 16 batch rows.
// Jump-head weights streamed block-wide by TDM (double-buffered 16KB chunks).
// ---------------------------------------------------------------------------
__global__ __launch_bounds__(128) void fused_kernel(
    const float* __restrict__ cur, const int* __restrict__ yidx,
    const char* __restrict__ ws, float* __restrict__ out,
    float* __restrict__ partials)
{
  const _Float16* W1D = (const _Float16*)(ws + OFF_W1D);
  const _Float16* W1J = (const _Float16*)(ws + OFF_W1J);
  const _Float16* W2D = (const _Float16*)(ws + OFF_W2D);
  const _Float16* W3D = (const _Float16*)(ws + OFF_W3D);
  const _Float16* W2J = (const _Float16*)(ws + OFF_W2J);
  const _Float16* W3J = (const _Float16*)(ws + OFF_W3J);
  const _Float16* W4D = (const _Float16*)(ws + OFF_W4D);
  const char*     W4J = ws + OFF_W4J;
  const float* B1D = (const float*)(ws + OFF_B1D);
  const float* B2D = (const float*)(ws + OFF_B2D);
  const float* B3D = (const float*)(ws + OFF_B3D);
  const float* B1J = (const float*)(ws + OFF_B1J);
  const float* B2J = (const float*)(ws + OFF_B2J);
  const float* B3J = (const float*)(ws + OFF_B3J);
  const float* B4D = (const float*)(ws + OFF_B4D);
  const float* B4J = (const float*)(ws + OFF_B4J);

  __shared__ __align__(1024) char smem[JOFF + 2 * CHBYTES];
  const int lane = threadIdx.x & 31;
  const int wid  = threadIdx.x >> 5;
  const int rowbase = blockIdx.x * (NWAVES * 16) + wid * 16;
  char* S0 = smem + (wid * 4 + 0) * SLOT;
  char* S1 = smem + (wid * 4 + 1) * SLOT;
  char* S2 = smem + (wid * 4 + 2) * SLOT;
  char* S3 = smem + (wid * 4 + 3) * SLOT;
  const int n  = lane & 15;
  const int kh = lane >> 4;
  const unsigned int ldsbase = (unsigned int)(size_t)(void*)smem;

  // Kick off TDM stream of the jump head early: chunks 0,1 -> double buffer.
  if (wid == 0) {
    tdm_load_1d(ldsbase + JOFF, W4J, CHBYTES);
    tdm_load_1d(ldsbase + JOFF + CHBYTES, W4J + CHBYTES, CHBYTES);
  }

  // ---- joint layer 1 of both towers (one pass over curstate) ----
  {
    v8f accD[8] = {};
    v8f accJ[8] = {};
#pragma unroll
    for (int kt = 0; kt < 8; ++kt) {
      v16h a = afrag_l1(cur, rowbase, lane, kt);
#pragma unroll
      for (int nt = 0; nt < 8; ++nt) {
        accD[nt] = wmma_f16(a, bfrag(W1D, nt * 8 + kt, lane), accD[nt]);
        accJ[nt] = wmma_f16(a, bfrag(W1J, nt * 8 + kt, lane), accJ[nt]);
      }
    }
    store_tanh(accD, B1D, (_Float16*)S0, lane);   // h1_d
    store_tanh(accJ, B1J, (_Float16*)S3, lane);   // h1_j
  }

  // ---- drift tower: L2 S0->S1, L3 (S1, res S0)->S2, head S2 -> drift f32 S1 ----
  { v8f acc[8] = {}; gemm_h((const _Float16*)S0, W2D, lane, acc); store_tanh(acc, B2D, (_Float16*)S1, lane); }
  { v8f acc[8] = {}; gemm_h((const _Float16*)S1, W3D, lane, acc);
    store_relures(acc, B3D, (const _Float16*)S0, (_Float16*)S2, lane); }
  {
    v8f acc[4] = {};
#pragma unroll
    for (int kt = 0; kt < 4; ++kt) {
      v16h a = afrag_lds((const _Float16*)S2, lane, kt);
#pragma unroll
      for (int nt = 0; nt < 4; ++nt)
        acc[nt] = wmma_f16(a, bfrag(W4D, nt * 4 + kt, lane), acc[nt]);
    }
    float* db = (float*)S1;
#pragma unroll
    for (int nt = 0; nt < 4; ++nt) {
      float bv = B4D[nt * 16 + n];
#pragma unroll
      for (int r = 0; r < 8; ++r)
        db[(r + 8 * kh) * FSTRIDE + nt * 16 + n] = acc[nt][r] + bv;
    }
  }

  // ---- jump tower: L2 S3->S0, L3 (S0, res S3)->S2 ----
  { v8f acc[8] = {}; gemm_h((const _Float16*)S3, W2J, lane, acc); store_tanh(acc, B2J, (_Float16*)S0, lane); }
  { v8f acc[8] = {}; gemm_h((const _Float16*)S0, W3J, lane, acc);
    store_relures(acc, B3J, (const _Float16*)S3, (_Float16*)S2, lane); }

  // ---- jump head: A resident in regs; stream 128 N-tiles from TDM-staged LDS ----
  v16h Aj[4];
#pragma unroll
  for (int kt = 0; kt < 4; ++kt) Aj[kt] = afrag_lds((const _Float16*)S2, lane, kt);

  int* ybuf = (int*)S0;                       // y tile [16][64]
  for (int i = lane; i < 16 * DIM; i += 32)
    ybuf[i] = yidx[((size_t)(rowbase + (i >> 6))) * DIM + (i & 63)];
  float* selbuf = (float*)S3;                 // sel f32 [16][FSTRIDE]

  float ent_acc = 0.f, lp_acc = 0.f;
  for (int ch = 0; ch < NCHUNK; ++ch) {
    if (wid == 0) {
      if (ch == NCHUNK - 1) __builtin_amdgcn_s_wait_tensorcnt(0);
      else                  __builtin_amdgcn_s_wait_tensorcnt(1);
    }
    __syncthreads();                           // chunk ch visible to all waves
    const _Float16* wb = (const _Float16*)(smem + JOFF + (size_t)(ch & 1) * CHBYTES);
#pragma unroll
    for (int tt = 0; tt < 4; ++tt) {
      const int t = ch * 4 + tt;
      v8f c = {};
#pragma unroll
      for (int kt = 0; kt < 4; ++kt)
        c = wmma_f16(Aj[kt], bfrag(wb, tt * 4 + kt, lane), c);
      float bv = B4J[t * 16 + n];

      if (t < 64) {
        // logits tile for dim d=t: 16 logits of a row live across a 16-lane group
#pragma unroll
        for (int r = 0; r < 8; ++r) {
          int m = r + 8 * kh;
          float v = c[r] + bv;
          float mx = v;
          mx = fmaxf(mx, __shfl_xor(mx, 1, 16));
          mx = fmaxf(mx, __shfl_xor(mx, 2, 16));
          mx = fmaxf(mx, __shfl_xor(mx, 4, 16));
          mx = fmaxf(mx, __shfl_xor(mx, 8, 16));
          float e = __expf(v - mx);
          float s = e;
          s += __shfl_xor(s, 1, 16);
          s += __shfl_xor(s, 2, 16);
          s += __shfl_xor(s, 4, 16);
          s += __shfl_xor(s, 8, 16);
          float lp = (v - mx) - __logf(s);
          ent_acc += (e / s) * lp;
          int yv = ybuf[m * 64 + t];
          if (n == yv) lp_acc += lp;
        }
      } else {
        int d = t - 64;                        // augmented jumps tile for dim d
#pragma unroll
        for (int r = 0; r < 8; ++r) {
          int m = r + 8 * kh;
          int yv = ybuf[m * 64 + d];
          if (n == yv) selbuf[m * FSTRIDE + d] = c[r] + bv;  // one lane per row
        }
      }
    }
    __syncthreads();                           // all waves done with buf (ch&1)
    if (wid == 0 && (ch + 2) < NCHUNK)
      tdm_load_1d(ldsbase + JOFF + (unsigned int)(ch & 1) * CHBYTES,
                  W4J + (size_t)(ch + 2) * CHBYTES, CHBYTES);
  }

  // ---- out = last + drift + sel ----
  {
    const float* db = (const float*)S1;
    for (int i = lane; i < 16 * DIM; i += 32) {
      int m = i >> 6, d = i & 63;
      size_t row = (size_t)(rowbase + m);
      out[row * DIM + d] = cur[((size_t)(PASTLEN - 1) * BATCH + row) * DIM + d]
                         + db[m * FSTRIDE + d] + selbuf[m * FSTRIDE + d];
    }
  }

  // ---- wave-reduce scalars, one slot per wave (deterministic) ----
#pragma unroll
  for (int off = 16; off; off >>= 1) {
    ent_acc += __shfl_xor(ent_acc, off, 32);
    lp_acc  += __shfl_xor(lp_acc, off, 32);
  }
  if (lane == 0) {
    int slot = blockIdx.x * NWAVES + wid;
    partials[2 * slot + 0] = ent_acc;
    partials[2 * slot + 1] = lp_acc;
  }
}

__global__ void finalize_kernel(const float* __restrict__ partials,
                                float* __restrict__ out) {
  if (threadIdx.x == 0 && blockIdx.x == 0) {
    float e = 0.f, l = 0.f;
    for (int i = 0; i < (BATCH / (NWAVES * 16)) * NWAVES; ++i) {
      e += partials[2 * i];
      l += partials[2 * i + 1];
    }
    out[(size_t)BATCH * DIM + 0] = e / (float)((size_t)BATCH * DIM);
    out[(size_t)BATCH * DIM + 1] = l;
  }
}

// ---------------------------------------------------------------------------
extern "C" void kernel_launch(void* const* d_in, const int* in_sizes, int n_in,
                              void* d_out, int out_size, void* d_ws, size_t ws_size,
                              hipStream_t stream) {
  (void)in_sizes; (void)n_in; (void)out_size;
  if (ws_size < WS_NEED) return;
  const float* cur  = (const float*)d_in[0];
  const int*   yidx = (const int*)d_in[1];
  const float* Wd1 = (const float*)d_in[2],  *bd1 = (const float*)d_in[3];
  const float* Wd2 = (const float*)d_in[4],  *bd2 = (const float*)d_in[5];
  const float* Wd3 = (const float*)d_in[6],  *bd3 = (const float*)d_in[7];
  const float* Wd4 = (const float*)d_in[8],  *bd4 = (const float*)d_in[9];
  const float* Wj1 = (const float*)d_in[10], *bj1 = (const float*)d_in[11];
  const float* Wj2 = (const float*)d_in[12], *bj2 = (const float*)d_in[13];
  const float* Wj3 = (const float*)d_in[14], *bj3 = (const float*)d_in[15];
  const float* Wj4 = (const float*)d_in[16], *bj4 = (const float*)d_in[17];
  char* ws = (char*)d_ws;

  prep_wtiles<<<(8 * 8 * 512 + 255) / 256, 256, 0, stream>>>(Wd1, (_Float16*)(ws + OFF_W1D), INDIM, HID, 8, 8);
  prep_wtiles<<<(8 * 8 * 512 + 255) / 256, 256, 0, stream>>>(Wj1, (_Float16*)(ws + OFF_W1J), INDIM, HID, 8, 8);
  prep_wtiles<<<(4 * 8 * 512 + 255) / 256, 256, 0, stream>>>(Wd2, (_Float16*)(ws + OFF_W2D), HID, HID, 4, 8);
  prep_wtiles<<<(4 * 8 * 512 + 255) / 256, 256, 0, stream>>>(Wd3, (_Float16*)(ws + OFF_W3D), HID, HID, 4, 8);
  prep_wtiles<<<(4 * 8 * 512 + 255) / 256, 256, 0, stream>>>(Wj2, (_Float16*)(ws + OFF_W2J), HID, HID, 4, 8);
  prep_wtiles<<<(4 * 8 * 512 + 255) / 256, 256, 0, stream>>>(Wj3, (_Float16*)(ws + OFF_W3J), HID, HID, 4, 8);
  prep_wtiles<<<(4 * 4 * 512 + 255) / 256, 256, 0, stream>>>(Wd4, (_Float16*)(ws + OFF_W4D), HID, DIM, 4, 4);
  prep_wj4<<<(128 * 4 * 512 + 255) / 256, 256, 0, stream>>>(Wj4, (_Float16*)(ws + OFF_W4J));

  prep_bias<<<1, HP, 0, stream>>>(bd1, (float*)(ws + OFF_B1D), HID, HP);
  prep_bias<<<1, HP, 0, stream>>>(bd2, (float*)(ws + OFF_B2D), HID, HP);
  prep_bias<<<1, HP, 0, stream>>>(bd3, (float*)(ws + OFF_B3D), HID, HP);
  prep_bias<<<1, HP, 0, stream>>>(bj1, (float*)(ws + OFF_B1J), HID, HP);
  prep_bias<<<1, HP, 0, stream>>>(bj2, (float*)(ws + OFF_B2J), HID, HP);
  prep_bias<<<1, HP, 0, stream>>>(bj3, (float*)(ws + OFF_B3J), HID, HP);
  prep_bias<<<1, DIM, 0, stream>>>(bd4, (float*)(ws + OFF_B4D), DIM, DIM);
  prep_bj4<<<JCOLS / 256, 256, 0, stream>>>(bj4, (float*)(ws + OFF_B4J));

  fused_kernel<<<BATCH / (NWAVES * 16), NWAVES * 32, 0, stream>>>(
      cur, yidx, (const char*)ws, (float*)d_out, (float*)(ws + OFF_PART));

  finalize_kernel<<<1, 32, 0, stream>>>((const float*)(ws + OFF_PART), (float*)d_out);
}
